// ELSA_84756884619417
// MI455X (gfx1250) — matrix operand order repaired and checked by reading, not compiled, not run on hardware
//
#include <hip/hip_runtime.h>

typedef __attribute__((ext_vector_type(16))) _Float16 v16h;
typedef __attribute__((ext_vector_type(2)))  _Float16 v2h;
typedef __attribute__((ext_vector_type(8)))  float    v8f;

#define NEG_SLOPE 0.2f
#define WLDS_STRIDE 132   // padded half-stride per output column to spread LDS banks

// ---------------------------------------------------------------------------
// helpers
// ---------------------------------------------------------------------------
__device__ inline unsigned keyOf(float v) {
  unsigned b = __float_as_uint(v);
  return b ^ ((b & 0x80000000u) ? 0xFFFFFFFFu : 0x80000000u);  // ascending-order key
}

__device__ inline void atomicMaxFloatBits(int* addr, float val) {
  int iv = __float_as_int(val);
  if (iv >= 0) atomicMax(addr, iv);
  else         atomicMin((unsigned int*)addr, (unsigned int)iv);
}

__device__ inline void edgeNodes(int e, int E, const int* __restrict__ src,
                                 const int* __restrict__ tgt, int& s, int& t) {
  if (e < E) { s = src[e]; t = tgt[e]; }
  else       { s = e - E;  t = e - E;  }  // self loops appended
}

// ---------------------------------------------------------------------------
// 0) init: zero output features, reset per-node reductions & select state
// ---------------------------------------------------------------------------
__global__ __launch_bounds__(256) void elsa_init_kernel(
    float* __restrict__ out_feat, int* __restrict__ mb, float* __restrict__ denom,
    float* __restrict__ asum, int* __restrict__ hist, int* __restrict__ state,
    int N, int K) {
  int idx = blockIdx.x * 256 + threadIdx.x;
  int total = N * 64;
  if (idx < total) out_feat[idx] = 0.0f;
  if (idx < N) {
    mb[idx]    = (int)0xFF800000;   // -inf bits
    denom[idx] = 0.0f;
    asum[idx]  = 0.0f;
  }
  if (idx < 256) hist[idx] = 0;
  if (idx == 0) { state[0] = 0; state[1] = K; state[2] = 0; }
}

// ---------------------------------------------------------------------------
// 1) h = x @ W  via v_wmma_f32_16x16x32_f16 (one 16-row tile per wave)
// ---------------------------------------------------------------------------
__global__ __launch_bounds__(256) void elsa_gemm_h_kernel(
    const float* __restrict__ x, const float* __restrict__ w,
    float* __restrict__ h, int N) {
  __shared__ _Float16 wlds[64 * WLDS_STRIDE];  // [n][k], padded

  // convert W (f32, [128][64]) -> f16 LDS transposed [64][128]
  for (int i = threadIdx.x; i < 128 * 64; i += 256) {
    int k = i >> 6;          // 0..127
    int n = i & 63;          // 0..63
    wlds[n * WLDS_STRIDE + k] = (_Float16)w[i];
  }
  __syncthreads();

  int wave = threadIdx.x >> 5;
  int lane = threadIdx.x & 31;
  int tile = blockIdx.x * 8 + wave;          // 16-row tile index
  int tiles = (N + 15) >> 4;
  if (tile >= tiles) return;
  int row_base = tile * 16;

  int hs = lane >> 4;        // half-select (0: lanes 0-15, 1: lanes 16-31)
  int m  = lane & 15;        // row within tile (A) / col within tile (B,C,D)
  const float* xrow = x + (size_t)(row_base + m) * 128;

  v8f acc0 = {}, acc1 = {}, acc2 = {}, acc3 = {};

#pragma unroll
  for (int kt = 0; kt < 4; ++kt) {
    // --- A fragment: 16x32 f16, ISA lane layout ---
    v16h a;
#pragma unroll
    for (int g = 0; g < 8; ++g) {
      int kb = kt * 32 + ((g < 4) ? 0 : 16) + hs * 8 + 2 * (g & 3);
      float2 xv = *(const float2*)(xrow + kb);
      a[2 * g]     = (_Float16)xv.x;
      a[2 * g + 1] = (_Float16)xv.y;
    }
    // --- 4 column tiles ---
#pragma unroll
    for (int nt = 0; nt < 4; ++nt) {
      int n = nt * 16 + m;
      v16h b;
#pragma unroll
      for (int g = 0; g < 8; ++g) {
        int k0 = kt * 32 + ((g < 4) ? 0 : 16) + hs * 8 + 2 * (g & 3);
        v2h bb = *(const v2h*)&wlds[n * WLDS_STRIDE + k0];
        b[2 * g]     = bb[0];
        b[2 * g + 1] = bb[1];
      }
      v8f& acc = (nt == 0) ? acc0 : (nt == 1) ? acc1 : (nt == 2) ? acc2 : acc3;
      acc = __builtin_amdgcn_wmma_f32_16x16x32_f16(
          false, a, false, b, (short)0, acc, false, false);
    }
  }

  // store D (8 VGPRs: lane half picks rows 0-7 / 8-15)
#pragma unroll
  for (int nt = 0; nt < 4; ++nt) {
    const v8f& acc = (nt == 0) ? acc0 : (nt == 1) ? acc1 : (nt == 2) ? acc2 : acc3;
#pragma unroll
    for (int r = 0; r < 8; ++r) {
      int row = row_base + r + hs * 8;
      h[(size_t)row * 64 + nt * 16 + m] = acc[r];
    }
  }
}

// ---------------------------------------------------------------------------
// 2) per-node attention dots: ai = h . att[:64], aj = h . att[64:]
// ---------------------------------------------------------------------------
__global__ __launch_bounds__(256) void elsa_attdot_kernel(
    const float* __restrict__ h, const float* __restrict__ att,
    float* __restrict__ ai, float* __restrict__ aj, int N) {
  int n = blockIdx.x * 256 + threadIdx.x;
  if (n >= N) return;
  const float* hp = h + (size_t)n * 64;
  float s0 = 0.0f, s1 = 0.0f;
#pragma unroll 8
  for (int c = 0; c < 64; ++c) {
    float v = hp[c];
    s0 += v * att[c];
    s1 += v * att[64 + c];
  }
  ai[n] = s0;
  aj[n] = s1;
}

// ---------------------------------------------------------------------------
// 3) edge logits + leaky relu + segment max over target
// ---------------------------------------------------------------------------
__global__ __launch_bounds__(256) void elsa_alpha_max_kernel(
    const float* __restrict__ ai, const float* __restrict__ aj,
    const int* __restrict__ src, const int* __restrict__ tgt,
    float* __restrict__ alpha, int* __restrict__ mb, int E, int EN) {
  int e = blockIdx.x * 256 + threadIdx.x;
  if (e >= EN) return;
  int s, t;
  edgeNodes(e, E, src, tgt, s, t);
  float a = ai[t] + aj[s];
  a = (a > 0.0f) ? a : NEG_SLOPE * a;
  alpha[e] = a;
  atomicMaxFloatBits(&mb[t], a);
}

// ---------------------------------------------------------------------------
// 4) a = exp(alpha - m[tgt]); denom[tgt] += a
// ---------------------------------------------------------------------------
__global__ __launch_bounds__(256) void elsa_exp_denom_kernel(
    float* __restrict__ alpha, const int* __restrict__ mb,
    float* __restrict__ denom, const int* __restrict__ src,
    const int* __restrict__ tgt, int E, int EN) {
  int e = blockIdx.x * 256 + threadIdx.x;
  if (e >= EN) return;
  int s, t;
  edgeNodes(e, E, src, tgt, s, t);
  float av = __expf(alpha[e] - __int_as_float(mb[t]));
  alpha[e] = av;
  atomicAdd(&denom[t], av);
}

// ---------------------------------------------------------------------------
// 5) alpha /= denom[tgt]; asum[src] += alpha
// ---------------------------------------------------------------------------
__global__ __launch_bounds__(256) void elsa_norm_kernel(
    float* __restrict__ alpha, const float* __restrict__ denom,
    float* __restrict__ asum, const int* __restrict__ src,
    const int* __restrict__ tgt, int E, int EN) {
  int e = blockIdx.x * 256 + threadIdx.x;
  if (e >= EN) return;
  int s, t;
  edgeNodes(e, E, src, tgt, s, t);
  float an = alpha[e] / (denom[t] + 1e-16f);
  alpha[e] = an;
  atomicAdd(&asum[s], an);
}

// ---------------------------------------------------------------------------
// 6) radix-select: per-pass 256-bin histogram of candidate keys
// ---------------------------------------------------------------------------
__global__ __launch_bounds__(256) void elsa_hist_kernel(
    const float* __restrict__ asum, const int* __restrict__ state,
    int* __restrict__ hist, int N, int shift) {
  __shared__ int lh[256];
  lh[threadIdx.x] = 0;
  __syncthreads();
  int n = blockIdx.x * 256 + threadIdx.x;
  if (n < N) {
    unsigned u = keyOf(asum[n]);
    unsigned prefix = (unsigned)state[0];
    bool cand = (shift == 24) || ((u >> (shift + 8)) == prefix);
    if (cand) atomicAdd(&lh[(u >> shift) & 255], 1);
  }
  __syncthreads();
  int v = lh[threadIdx.x];
  if (v) atomicAdd(&hist[threadIdx.x], v);
}

// single-block: pick the bin where cumulative (from top) crosses `remaining`
__global__ __launch_bounds__(256) void elsa_scan_kernel(
    int* __restrict__ state, int* __restrict__ hist) {
  if (threadIdx.x == 0) {
    int remaining = state[1];
    int cum = 0, b = 0;
    for (int j = 255; j >= 0; --j) {
      int hj = hist[j];
      if (cum + hj >= remaining) { b = j; break; }
      cum += hj;
    }
    state[0] = (state[0] << 8) | b;
    state[1] = remaining - cum;
  }
  __syncthreads();
  hist[threadIdx.x] = 0;  // reset for next pass
}

// ---------------------------------------------------------------------------
// 7) final selection: key > T, plus ties up to quota
// ---------------------------------------------------------------------------
__global__ __launch_bounds__(256) void elsa_select_kernel(
    const float* __restrict__ asum, int* __restrict__ state,
    int* __restrict__ nmask, float* __restrict__ nmask_out, int N) {
  int n = blockIdx.x * 256 + threadIdx.x;
  if (n >= N) return;
  unsigned T = (unsigned)state[0];
  int quota = state[1];
  unsigned u = keyOf(asum[n]);
  int m = 0;
  if (u > T) {
    m = 1;
  } else if (u == T) {
    int slot = atomicAdd(&state[2], 1);
    m = (slot < quota) ? 1 : 0;
  }
  nmask[n] = m;
  nmask_out[n] = (float)m;
}

// ---------------------------------------------------------------------------
// 8) edge mask + weighted scatter-add aggregation (4 cols per thread)
// ---------------------------------------------------------------------------
__global__ __launch_bounds__(256) void elsa_aggregate_kernel(
    const float* __restrict__ h, const float* __restrict__ alpha,
    const int* __restrict__ src, const int* __restrict__ tgt,
    const int* __restrict__ nmask, float* __restrict__ out_feat,
    float* __restrict__ emask_out, int E, int EN) {
  int tid = blockIdx.x * 256 + threadIdx.x;
  int e = tid >> 4;
  if (e >= EN) return;
  int chunk = tid & 15;
  int s, t;
  edgeNodes(e, E, src, tgt, s, t);
  int em = nmask[s] & nmask[t];
  if (chunk == 0) emask_out[e] = em ? 1.0f : 0.0f;
  if (!em) return;
  float wgt = alpha[e];
  int c = chunk * 4;
  float4 hv = *(const float4*)(h + (size_t)s * 64 + c);
  float* op = out_feat + (size_t)t * 64 + c;
  atomicAdd(op + 0, wgt * hv.x);
  atomicAdd(op + 1, wgt * hv.y);
  atomicAdd(op + 2, wgt * hv.z);
  atomicAdd(op + 3, wgt * hv.w);
}

// ---------------------------------------------------------------------------
// launch
// ---------------------------------------------------------------------------
extern "C" void kernel_launch(void* const* d_in, const int* in_sizes, int n_in,
                              void* d_out, int out_size, void* d_ws, size_t ws_size,
                              hipStream_t stream) {
  const float* x   = (const float*)d_in[0];
  const float* w   = (const float*)d_in[1];
  const float* att = (const float*)d_in[2];
  const int*   ei  = (const int*)d_in[3];

  int N  = in_sizes[0] / 128;   // IN_C = 128
  int E  = in_sizes[3] / 2;     // edge_index is [2, E]
  int EN = E + N;               // with self loops
  int K  = (N + 1) / 2;         // ceil(RATIO * N), RATIO = 0.5

  const int* src = ei;
  const int* tgt = ei + E;

  // workspace carve-up
  float* h     = (float*)d_ws;               // N*64
  float* ai    = h + (size_t)N * 64;         // N
  float* aj    = ai + N;                     // N
  int*   mb    = (int*)(aj + N);             // N (float bits, segment max)
  float* denom = (float*)(mb + N);           // N
  float* asum  = denom + N;                  // N
  float* alpha = asum + N;                   // EN
  int*   nmask = (int*)(alpha + EN);         // N
  int*   hist  = nmask + N;                  // 256
  int*   state = hist + 256;                 // [prefix, remaining, tie_counter]

  // outputs: [out (N*64) | edge_mask (EN) | node_mask (N)]
  float* out_feat  = (float*)d_out;
  float* emask_out = out_feat + (size_t)N * 64;
  float* nmask_out = emask_out + EN;

  int tiles = (N + 15) / 16;

  elsa_init_kernel<<<(N * 64 + 255) / 256, 256, 0, stream>>>(
      out_feat, mb, denom, asum, hist, state, N, K);
  elsa_gemm_h_kernel<<<(tiles + 7) / 8, 256, 0, stream>>>(x, w, h, N);
  elsa_attdot_kernel<<<(N + 255) / 256, 256, 0, stream>>>(h, att, ai, aj, N);
  elsa_alpha_max_kernel<<<(EN + 255) / 256, 256, 0, stream>>>(
      ai, aj, src, tgt, alpha, mb, E, EN);
  elsa_exp_denom_kernel<<<(EN + 255) / 256, 256, 0, stream>>>(
      alpha, mb, denom, src, tgt, E, EN);
  elsa_norm_kernel<<<(EN + 255) / 256, 256, 0, stream>>>(
      alpha, denom, asum, src, tgt, E, EN);

  for (int p = 0; p < 4; ++p) {
    elsa_hist_kernel<<<(N + 255) / 256, 256, 0, stream>>>(
        asum, state, hist, N, 24 - 8 * p);
    elsa_scan_kernel<<<1, 256, 0, stream>>>(state, hist);
  }

  elsa_select_kernel<<<(N + 255) / 256, 256, 0, stream>>>(
      asum, state, nmask, nmask_out, N);

  int agg_threads = EN * 16;
  elsa_aggregate_kernel<<<(agg_threads + 255) / 256, 256, 0, stream>>>(
      h, alpha, src, tgt, nmask, out_feat, emask_out, E, EN);
}
